// SerializedAttention_13529146982501
// MI455X (gfx1250) — matrix operand order repaired and checked by reading, not compiled, not run on hardware
//
#include <hip/hip_runtime.h>
#include <hip/hip_bf16.h>

typedef __attribute__((ext_vector_type(16))) __bf16 v16bf;
typedef __attribute__((ext_vector_type(8)))  float  v8f;

static constexpr int Npts = 65536;
static constexpr int C    = 512;
static constexpr int H    = 8;
static constexpr int DH   = 64;
static constexpr int NW   = 512;     // windows = N / 128
static constexpr int RPE_NUM = 31;
static constexpr float QK_SCALE = 0.125f;  // DH^-0.5

// ---- CDNA5 async global->LDS path (ASYNCcnt), with portable fallback -------
#if defined(__has_builtin)
#  if __has_builtin(__builtin_amdgcn_global_load_async_to_lds_b128)
#    define ASYNC_LDS 1
#  endif
#endif
#ifndef ASYNC_LDS
#  define ASYNC_LDS 0
#endif

typedef int v4i_vs __attribute__((vector_size(16)));   // matches builtin's V4i

__device__ __forceinline__ void cp16(const unsigned short* g, unsigned short* l) {
#if ASYNC_LDS
  __builtin_amdgcn_global_load_async_to_lds_b128(
      (__attribute__((address_space(1))) v4i_vs*)g,
      (__attribute__((address_space(3))) v4i_vs*)l, 0, 0);
#else
  *(uint4*)l = *(const uint4*)g;
#endif
}
__device__ __forceinline__ void async_wait() {
#if ASYNC_LDS
  asm volatile("s_wait_asynccnt 0" ::: "memory");
#endif
}

__device__ __forceinline__ unsigned short f2bf(float x) {
  union { float f; unsigned int u; } c; c.f = x;
  unsigned int r = c.u + 0x7FFFu + ((c.u >> 16) & 1u);   // round-to-nearest-even
  return (unsigned short)(r >> 16);
}

union FragB16 { v16bf v; unsigned int u[8]; unsigned short h[16]; };

// A-matrix fragment (16x32 bf16). lds is row-major [M][K], strideHalf in bf16 units.
// Layout: lane<16 holds M=lane, K=0..15; lane>=16 K=16..31 (matches frag_b).
__device__ __forceinline__ v16bf frag_a(const unsigned short* lds, int strideHalf,
                                        int m0, int kbase, int lane) {
  FragB16 f;
  const unsigned int* p =
      (const unsigned int*)(lds + (size_t)(m0 + (lane & 15)) * strideHalf + kbase);
  const int hi = (lane >> 4) << 3;   // +8 dwords (K+16) for upper half-wave
#pragma unroll
  for (int j = 0; j < 8; ++j) f.u[j] = p[j + hi];
  return f.v;
}

// B-matrix fragment (32x16 bf16), B[k][n] = lds[n][k] (lds row-major [N][K]).
// Layout: lanes 0-15 hold K=0..15 (N=lane), lanes 16-31 hold K=16..31.
__device__ __forceinline__ v16bf frag_b(const unsigned short* lds, int strideHalf,
                                        int n0, int kbase, int lane) {
  FragB16 f;
  const unsigned int* p =
      (const unsigned int*)(lds + (size_t)(n0 + (lane & 15)) * strideHalf + kbase);
  const int hi = (lane >> 4) << 3;   // +8 dwords (K+16) for upper half-wave
#pragma unroll
  for (int j = 0; j < 8; ++j) f.u[j] = p[j + hi];
  return f.v;
}

__global__ void f32_to_bf16(const float* __restrict__ in,
                            unsigned short* __restrict__ out, int n) {
  int i = blockIdx.x * blockDim.x + threadIdx.x;
  if (i < n) out[i] = f2bf(in[i]);
}

// out[n,m] = sum_k A[n,k] * Wt[m,k] + bias[m].
// Block tile 256x128, 8 waves, wave tile 32x128 (2x8 C-tiles), K sliced by 64.
// A frags reused across 8 N-tiles; B frags reused across 2 M-subtiles.
template <bool OUT_BF16>
__global__ __launch_bounds__(256) void gemm_bf16_wmma(
    const unsigned short* __restrict__ A,   // [Nrows, K] bf16
    const unsigned short* __restrict__ Wt,  // [Mout, K] bf16
    const float* __restrict__ bias,         // [Mout]
    void* __restrict__ outp,                // [Nrows, Mout]
    int K, int Mout) {
  __shared__ __align__(16) unsigned short lA[256 * 64];   // 32 KB
  __shared__ __align__(16) unsigned short lB[128 * 64];   // 16 KB
  const int tid = threadIdx.x;
  const int wave = tid >> 5, lane = tid & 31;
  const int row0 = blockIdx.y * 256;
  const int col0 = blockIdx.x * 128;

  v8f acc[2][8] = {};

  const int rb = tid >> 1, ob = (tid & 1) * 32;   // B: half row per thread

  for (int k0 = 0; k0 < K; k0 += 64) {
    __syncthreads();
    {
      const unsigned short* ga = A + (size_t)(row0 + tid) * K + k0;  // full row
      unsigned short* la = lA + tid * 64;
#pragma unroll
      for (int i = 0; i < 8; ++i) cp16(ga + i * 8, la + i * 8);
      const unsigned short* gb = Wt + (size_t)(col0 + rb) * K + k0 + ob;
      unsigned short* lb = lB + rb * 64 + ob;
#pragma unroll
      for (int i = 0; i < 4; ++i) cp16(gb + i * 8, lb + i * 8);
      if (k0 + 64 < K) {
        __builtin_prefetch(ga + 64, 0, 1);
        __builtin_prefetch(gb + 64, 0, 1);
      }
    }
    async_wait();
    __syncthreads();

    v16bf a[2][2];
#pragma unroll
    for (int mt = 0; mt < 2; ++mt)
#pragma unroll
      for (int kc = 0; kc < 2; ++kc)
        a[mt][kc] = frag_a(lA, 64, wave * 32 + mt * 16, kc * 32, lane);
#pragma unroll
    for (int j = 0; j < 8; ++j) {
#pragma unroll
      for (int kc = 0; kc < 2; ++kc) {
        v16bf b = frag_b(lB, 64, j * 16, kc * 32, lane);
#pragma unroll
        for (int mt = 0; mt < 2; ++mt)
          acc[mt][j] = __builtin_amdgcn_wmma_f32_16x16x32_bf16(
              false, a[mt][kc], false, b, (short)0, acc[mt][j], false, false);
      }
    }
  }

#pragma unroll
  for (int mt = 0; mt < 2; ++mt) {
    const int mrow = row0 + wave * 32 + mt * 16 + ((lane >> 4) << 3);
#pragma unroll
    for (int j = 0; j < 8; ++j) {
      const int n = col0 + j * 16 + (lane & 15);
      const float bn = bias[n];
#pragma unroll
      for (int v = 0; v < 8; ++v) {
        const float val = acc[mt][j][v] + bn;
        const size_t idx = (size_t)(mrow + v) * Mout + n;
        if (OUT_BF16) ((unsigned short*)outp)[idx] = f2bf(val);
        else          ((float*)outp)[idx]          = val;
      }
    }
  }
}

// One block per (window, head). Gather via `order`, S=QK^T in register tiles,
// RPE bias + exact softmax (half-wave shfl reductions), P->LDS bf16 (aliases
// dead Q/K), O=PV, scatter back through `order` (== out[inverse]).
__global__ __launch_bounds__(256) void attn_wmma(
    const unsigned short* __restrict__ qkv,
    const int* __restrict__ order,
    const int* __restrict__ gcoord,     // [N,3]
    const float* __restrict__ rpe,      // [93, 8]
    unsigned short* __restrict__ outB)  // [N, C] bf16, point order
{
  __shared__ __align__(16) char smem[54176];
  unsigned short* lQ  = (unsigned short*)smem;        // 128x64 bf16
  unsigned short* lK  = lQ + 128 * 64;                // 128x64 bf16
  unsigned short* lP  = lQ;                           // 128x128 bf16 (aliases Q+K)
  unsigned short* lVt = lK + 128 * 64;                // 64x128 bf16: lVt[d][q]
  int*   lOrd = (int*)(smem + 49152);                 // 128
  int*   lGc  = (int*)(smem + 49664);                 // 128x3
  float* lRpe = (float*)(smem + 51200);               // 93x8

  const int w = blockIdx.x, h = blockIdx.y;
  const int tid = threadIdx.x, wave = tid >> 5, lane = tid & 31;

  if (tid < 128) lOrd[tid] = order[w * 128 + tid];
  for (int i = tid; i < 3 * RPE_NUM * H; i += 256) lRpe[i] = rpe[i];
  __syncthreads();
  if (tid < 128) {
    const int n = lOrd[tid];
    lGc[tid * 3 + 0] = gcoord[n * 3 + 0];
    lGc[tid * 3 + 1] = gcoord[n * 3 + 1];
    lGc[tid * 3 + 2] = gcoord[n * 3 + 2];
  }
  {
    const int p = tid >> 1, half = (tid & 1) * 32;     // 2 threads per point row
    const size_t base = (size_t)lOrd[p] * (3 * C) + (size_t)h * DH + half;
    const unsigned short* gq = qkv + base;
    unsigned short* dq = lQ + p * 64 + half;
#pragma unroll
    for (int i = 0; i < 4; ++i) cp16(gq + i * 8, dq + i * 8);
    const unsigned short* gk = qkv + base + C;
    unsigned short* dk = lK + p * 64 + half;
#pragma unroll
    for (int i = 0; i < 4; ++i) cp16(gk + i * 8, dk + i * 8);
    union { uint4 q[4]; unsigned short s[32]; } vb;    // V through VGPRs (transpose)
    const uint4* gv = (const uint4*)(qkv + base + 2 * C);
    vb.q[0] = gv[0]; vb.q[1] = gv[1]; vb.q[2] = gv[2]; vb.q[3] = gv[3];
#pragma unroll
    for (int d = 0; d < 32; ++d) lVt[(half + d) * 128 + p] = vb.s[d];
  }
  async_wait();
  __syncthreads();

  // ---- S = Q K^T : wave owns rows [wave*16, wave*16+16), 8 N-tiles, K=64=2x32
  v8f acc[8] = {};
  {
    v16bf a0 = frag_a(lQ, 64, wave * 16, 0, lane);
    v16bf a1 = frag_a(lQ, 64, wave * 16, 32, lane);
#pragma unroll
    for (int j = 0; j < 8; ++j) {
      v16bf b0 = frag_b(lK, 64, j * 16, 0, lane);
      acc[j] = __builtin_amdgcn_wmma_f32_16x16x32_bf16(
          false, a0, false, b0, (short)0, acc[j], false, false);
      v16bf b1 = frag_b(lK, 64, j * 16, 32, lane);
      acc[j] = __builtin_amdgcn_wmma_f32_16x16x32_bf16(
          false, a1, false, b1, (short)0, acc[j], false, false);
    }
  }

  // ---- scale + RPE bias, per C-layout element (m = mbase+v, q = j*16+lane&15)
  const int mbase = wave * 16 + ((lane >> 4) << 3);
  int gmx[8], gmy[8], gmz[8];
#pragma unroll
  for (int v = 0; v < 8; ++v) {
    gmx[v] = lGc[(mbase + v) * 3 + 0];
    gmy[v] = lGc[(mbase + v) * 3 + 1];
    gmz[v] = lGc[(mbase + v) * 3 + 2];
  }
#pragma unroll
  for (int j = 0; j < 8; ++j) {
    const int q = j * 16 + (lane & 15);
    const int qx = lGc[q * 3 + 0], qy = lGc[q * 3 + 1], qz = lGc[q * 3 + 2];
#pragma unroll
    for (int v = 0; v < 8; ++v) {
      const int i0 = min(max(gmx[v] - qx, -15), 15) + 15;        // rows 0..30
      const int i1 = min(max(gmy[v] - qy, -15), 15) + 15 + 31;   // +RPE_NUM
      const int i2 = min(max(gmz[v] - qz, -15), 15) + 15 + 62;   // +2*RPE_NUM
      acc[j][v] = acc[j][v] * QK_SCALE +
                  lRpe[i0 * H + h] + lRpe[i1 * H + h] + lRpe[i2 * H + h];
    }
  }

  // ---- exact softmax per row: rows live inside one 16-lane half of the wave
  float invs[8];
#pragma unroll
  for (int v = 0; v < 8; ++v) {
    float m = acc[0][v];
#pragma unroll
    for (int j = 1; j < 8; ++j) m = fmaxf(m, acc[j][v]);
    m = fmaxf(m, __shfl_xor(m, 1));
    m = fmaxf(m, __shfl_xor(m, 2));
    m = fmaxf(m, __shfl_xor(m, 4));
    m = fmaxf(m, __shfl_xor(m, 8));
    float s = 0.f;
#pragma unroll
    for (int j = 0; j < 8; ++j) { float e = __expf(acc[j][v] - m); acc[j][v] = e; s += e; }
    s += __shfl_xor(s, 1);
    s += __shfl_xor(s, 2);
    s += __shfl_xor(s, 4);
    s += __shfl_xor(s, 8);
    invs[v] = 1.0f / s;
  }

  __syncthreads();   // all waves done reading lQ/lK -> safe to overwrite with P
#pragma unroll
  for (int j = 0; j < 8; ++j) {
    const int q = j * 16 + (lane & 15);
#pragma unroll
    for (int v = 0; v < 8; ++v)
      lP[(mbase + v) * 128 + q] = f2bf(acc[j][v] * invs[v]);
  }
  __syncthreads();

  // ---- O = P V : 4 DH-tiles, K=128=4x32; B[k=q][n=d] = lVt[d][q]
  v16bf ap[4];
#pragma unroll
  for (int kc = 0; kc < 4; ++kc) ap[kc] = frag_a(lP, 128, wave * 16, kc * 32, lane);
#pragma unroll
  for (int t = 0; t < 4; ++t) {
    v8f o = {};
#pragma unroll
    for (int kc = 0; kc < 4; ++kc) {
      v16bf bv = frag_b(lVt, 128, t * 16, kc * 32, lane);
      o = __builtin_amdgcn_wmma_f32_16x16x32_bf16(
          false, ap[kc], false, bv, (short)0, o, false, false);
    }
    const int d = t * 16 + (lane & 15);
#pragma unroll
    for (int v = 0; v < 8; ++v) {
      const int n = lOrd[mbase + v];
      outB[(size_t)n * C + h * DH + d] = f2bf(o[v]);
    }
  }
}

extern "C" void kernel_launch(void* const* d_in, const int* in_sizes, int n_in,
                              void* d_out, int out_size, void* d_ws, size_t ws_size,
                              hipStream_t stream) {
  const float* feat   = (const float*)d_in[0];
  const float* w_qkv  = (const float*)d_in[1];
  const float* b_qkv  = (const float*)d_in[2];
  const float* w_proj = (const float*)d_in[3];
  const float* b_proj = (const float*)d_in[4];
  const float* rpe    = (const float*)d_in[5];
  const int*   gcoord = (const int*)d_in[6];
  const int*   order  = (const int*)d_in[7];
  // d_in[8] = inverse : unused (we scatter through `order` instead)

  // workspace carve (bf16 buffers): feat | w_qkv | w_proj | qkv | attn_out
  unsigned short* featB  = (unsigned short*)d_ws;
  unsigned short* wqkvB  = featB  + (size_t)Npts * C;
  unsigned short* wprojB = wqkvB  + (size_t)3 * C * C;
  unsigned short* qkvB   = wprojB + (size_t)C * C;
  unsigned short* attnB  = qkvB   + (size_t)Npts * 3 * C;

  f32_to_bf16<<<(Npts * C + 255) / 256, 256, 0, stream>>>(feat, featB, Npts * C);
  f32_to_bf16<<<(3 * C * C + 255) / 256, 256, 0, stream>>>(w_qkv, wqkvB, 3 * C * C);
  f32_to_bf16<<<(C * C + 255) / 256, 256, 0, stream>>>(w_proj, wprojB, C * C);

  // QKV projection: [N,512] x [1536,512]^T -> [N,1536] bf16
  gemm_bf16_wmma<true><<<dim3(3 * C / 128, Npts / 256), 256, 0, stream>>>(
      featB, wqkvB, b_qkv, qkvB, C, 3 * C);

  // fused windowed attention (gather/scatter through `order`)
  attn_wmma<<<dim3(NW, H), 256, 0, stream>>>(qkvB, order, gcoord, rpe, attnB);

  // output projection: [N,512] x [512,512]^T -> d_out fp32
  gemm_bf16_wmma<false><<<dim3(C / 128, Npts / 256), 256, 0, stream>>>(
      attnB, wprojB, b_proj, d_out, C, C);
}
